// PoseTransformer_7576322310570
// MI455X (gfx1250) — matrix operand order repaired and checked by reading, not compiled, not run on hardware
//
#include <hip/hip_runtime.h>
#include <hip/hip_bf16.h>
#include <hip/hip_fp16.h>
#include <stdint.h>

// ---------------------------------------------------------------------------
// CDNA5 (gfx1250) PoseTransformer forward.
// - All dense linears run on v_wmma_f32_16x16x32_f16 (wave32, f32 accum).
// - Weights repacked once per launch into WMMA B-fragment order (f16).
// - Activations feeding GEMMs are produced directly in f16; A tiles are
//   copied global->LDS with GLOBAL_LOAD_ASYNC_TO_LDS_B128 (ASYNCcnt),
//   double-buffered so the copy of chunk k+1 overlaps WMMAs of chunk k.
// - Irregular work (grid_sample, 17-token attention, layernorm, offsets)
//   stays in scalar VALU kernels (<5% of FLOPs).
// ---------------------------------------------------------------------------

typedef __attribute__((ext_vector_type(16))) _Float16 v16h;
typedef __attribute__((ext_vector_type(8)))  float    v8f;

#define NB   256      // batch
#define NP   17       // keypoints
#define B17  4352     // NB*NP tokens
#define TL   17408    // NB*4*NP deformable tokens
#define DJNT 768

__device__ __forceinline__ float gelu_f(float x) {
    return 0.5f * x * (1.0f + erff(x * 0.70710678118654752f));
}

// ---------------------------------------------------------------------------
// Weight packing: f32 (K x N row-major) -> f16 dwords in WMMA B-fragment
// order for 16x16x32 tiles. Packed dword index:
//   (((ntile*(K/32) + kc)*32 + lane)*8 + v)
// holds { W[k0,n], W[k0+1,n] } with n = ntile*16 + (lane&15),
// k0 = kc*32 + (lane>>4)*16 + 2v.
// ---------------------------------------------------------------------------
__global__ void pack_weight_k(const float* __restrict__ W,
                              uint32_t* __restrict__ out, int K, int N) {
    long total = (long)K * N / 2;
    long t = (long)blockIdx.x * 256 + threadIdx.x;
    if (t >= total) return;
    int v = (int)(t & 7);
    long r = t >> 3;
    int lane = (int)(r & 31);
    r >>= 5;
    int KC = K >> 5;
    int kc = (int)(r % KC);
    int ntile = (int)(r / KC);
    int n  = ntile * 16 + (lane & 15);
    int k0 = kc * 32 + ((lane >> 4) << 4) + (v << 1);
    union { uint32_t u; _Float16 h[2]; } pk;
    pk.h[0] = (_Float16)W[(long)k0 * N + n];
    pk.h[1] = (_Float16)W[(long)(k0 + 1) * N + n];
    out[t] = pk.u;
}

// ---------------------------------------------------------------------------
// WMMA GEMM: out[M,N] = epilogue(A[M,K](f16) @ Wpacked + bias [+ residual])
// Block = 4 waves (128 thr); block tile = 32(M) x 128(N); each wave computes
// a 32x32 tile: 2 shared A fragments x 2 B fragments -> 4 WMMAs per K-chunk.
// A tiles (32x32 f16 = 2KB) are fetched with async global->LDS b128 copies,
// double buffered. Requires M%32==0, K%32==0, N%128==0.
// act: 1 = exact GELU.  Ch!=null -> f16 output (no residual), else f32 +res.
// ---------------------------------------------------------------------------
__global__ __launch_bounds__(128) void gemm_wmma_f16(
    const _Float16* __restrict__ A, const uint32_t* __restrict__ Wp,
    const float* __restrict__ bias, const float* res,
    float* Cf, _Float16* Ch, int M, int K, int N, int act)
{
    __shared__ __align__(16) uint32_t ldsA[1024];   // 2 buffers x 32x32 f16
    const int lane  = threadIdx.x & 31;
    const int wave  = threadIdx.x >> 5;
    const int mbase = blockIdx.x * 32;
    const int ntb   = blockIdx.y * 8 + wave * 2;    // 2 n-tiles per wave
    const int KC    = K >> 5;

    // async-copy addressing for this thread (one b128 per thread per chunk)
    const int cp_row = threadIdx.x >> 2;            // 0..31
    const int cp_seg = threadIdx.x & 3;             // 16B segment in 64B row
    const uint64_t ga_row = (uint64_t)(uintptr_t)A +
        ((uint64_t)(mbase + cp_row) * (uint64_t)K) * 2 + (uint64_t)cp_seg * 16;
    const uint32_t lds_base = (uint32_t)(uintptr_t)(&ldsA[0]);
    const uint32_t lds_cp   = lds_base + (uint32_t)(cp_row * 64 + cp_seg * 16);

    union U16 { v16h v; uint32_t d[8]; uint4 q[2]; };
    U16 a0, a1, b0, b1;
    v8f c00 = {}, c01 = {}, c10 = {}, c11 = {};

    const uint4* wq0 = (const uint4*)(Wp + (((long)(ntb + 0) * KC) * 32 + lane) * 8);
    const uint4* wq1 = (const uint4*)(Wp + (((long)(ntb + 1) * KC) * 32 + lane) * 8);

    // A-fragment LDS read addresses (row-major f16 tile, 16 dwords per row)
    const int m0   = lane & 15;
    const int half = lane >> 4;
    const uint32_t* fr0 = ldsA + (m0 * 16 + half * 8);        // rows 0..15
    const uint32_t* fr1 = fr0 + 256;                          // rows 16..31

    // prologue: issue copy of chunk 0 into buffer 0
    {
        uint64_t ga = ga_row;                                 // kc = 0
        asm volatile("global_load_async_to_lds_b128 %0, %1, off"
                     :: "v"(lds_cp), "v"(ga) : "memory");
    }

    for (int kc = 0; kc < KC; ++kc) {
        // wait for my async copies, then cross-wave visibility
        asm volatile("s_wait_asynccnt 0x0" ::: "memory");
        __syncthreads();
        // overlap: issue copy of chunk kc+1 into the other buffer
        if (kc + 1 < KC) {
            uint64_t ga = ga_row + (uint64_t)(kc + 1) * 64;
            uint32_t ld = lds_cp + (uint32_t)(((kc + 1) & 1) * 2048);
            asm volatile("global_load_async_to_lds_b128 %0, %1, off"
                         :: "v"(ld), "v"(ga) : "memory");
        }
        // A fragments from current buffer
        const uint4* aq0 = (const uint4*)(fr0 + (kc & 1) * 512);
        const uint4* aq1 = (const uint4*)(fr1 + (kc & 1) * 512);
        a0.q[0] = aq0[0]; a0.q[1] = aq0[1];
        a1.q[0] = aq1[0]; a1.q[1] = aq1[1];
        // B fragments (pre-packed, lane-contiguous, L2 resident)
        b0.q[0] = wq0[0]; b0.q[1] = wq0[1]; wq0 += 64;
        b1.q[0] = wq1[0]; b1.q[1] = wq1[1]; wq1 += 64;
        c00 = __builtin_amdgcn_wmma_f32_16x16x32_f16(false, a0.v, false, b0.v,
                                                     (short)0, c00, false, false);
        c01 = __builtin_amdgcn_wmma_f32_16x16x32_f16(false, a0.v, false, b1.v,
                                                     (short)0, c01, false, false);
        c10 = __builtin_amdgcn_wmma_f32_16x16x32_f16(false, a1.v, false, b0.v,
                                                     (short)0, c10, false, false);
        c11 = __builtin_amdgcn_wmma_f32_16x16x32_f16(false, a1.v, false, b1.v,
                                                     (short)0, c11, false, false);
        // all waves done reading current buffer before it is overwritten
        __syncthreads();
    }

    const int rh = (lane >> 4) << 3;
#pragma unroll
    for (int t = 0; t < 4; ++t) {
        v8f c = (t == 0) ? c00 : (t == 1) ? c01 : (t == 2) ? c10 : c11;
        int ncol = (ntb + (t & 1)) * 16 + (lane & 15);
        int rbase = mbase + ((t >> 1) << 4) + rh;
        float bcol = bias[ncol];
#pragma unroll
        for (int v = 0; v < 8; ++v) {
            int row = rbase + v;
            float x = c[v] + bcol;
            if (act) x = gelu_f(x);
            long idx = (long)row * N + ncol;
            if (Ch) {
                Ch[idx] = (_Float16)x;
            } else {
                if (res) x += res[idx];
                Cf[idx] = x;
            }
        }
    }
}

// ---------------------------------------------------------------------------
// LayerNorm over last dim D (one token per 128-thread block). f32 output.
// ---------------------------------------------------------------------------
__global__ __launch_bounds__(128) void layernorm_k(
    const float* __restrict__ X, const float* __restrict__ G,
    const float* __restrict__ Bt, float* __restrict__ Y, int D)
{
    __shared__ float sm[128];
    long t = blockIdx.x;
    const float* x = X + t * D;
    int tid = threadIdx.x;
    float s = 0.f;
    for (int j = tid; j < D; j += 128) s += x[j];
    sm[tid] = s; __syncthreads();
    for (int o = 64; o > 0; o >>= 1) { if (tid < o) sm[tid] += sm[tid + o]; __syncthreads(); }
    float mean = sm[0] / (float)D; __syncthreads();
    float s2 = 0.f;
    for (int j = tid; j < D; j += 128) { float d = x[j] - mean; s2 += d * d; }
    sm[tid] = s2; __syncthreads();
    for (int o = 64; o > 0; o >>= 1) { if (tid < o) sm[tid] += sm[tid + o]; __syncthreads(); }
    float rstd = rsqrtf(sm[0] / (float)D + 1e-6f);
    float* y = Y + t * D;
    for (int j = tid; j < D; j += 128) y[j] = (x[j] - mean) * rstd * G[j] + Bt[j];
}

// Same, but f16 output (feeds WMMA GEMM A operand).
__global__ __launch_bounds__(128) void layernorm16_k(
    const float* __restrict__ X, const float* __restrict__ G,
    const float* __restrict__ Bt, _Float16* __restrict__ Y, int D)
{
    __shared__ float sm[128];
    long t = blockIdx.x;
    const float* x = X + t * D;
    int tid = threadIdx.x;
    float s = 0.f;
    for (int j = tid; j < D; j += 128) s += x[j];
    sm[tid] = s; __syncthreads();
    for (int o = 64; o > 0; o >>= 1) { if (tid < o) sm[tid] += sm[tid + o]; __syncthreads(); }
    float mean = sm[0] / (float)D; __syncthreads();
    float s2 = 0.f;
    for (int j = tid; j < D; j += 128) { float d = x[j] - mean; s2 += d * d; }
    sm[tid] = s2; __syncthreads();
    for (int o = 64; o > 0; o >>= 1) { if (tid < o) sm[tid] += sm[tid + o]; __syncthreads(); }
    float rstd = rsqrtf(sm[0] / (float)D + 1e-6f);
    _Float16* y = Y + t * D;
    for (int j = tid; j < D; j += 128)
        y[j] = (_Float16)((x[j] - mean) * rstd * G[j] + Bt[j]);
}

// LayerNorm of (XL[t] + X0[b,p]) for deformable blocks, D = 128, f32 out.
__global__ __launch_bounds__(128) void ln_add0_k(
    const float* __restrict__ XL, const float* __restrict__ X0,
    const float* __restrict__ G, const float* __restrict__ Bt,
    float* __restrict__ Y)
{
    __shared__ float sm[128];
    long t = blockIdx.x;
    int p = (int)(t % NP);
    long b = t / (4 * NP);
    int tid = threadIdx.x;
    float v = XL[t * 128 + tid] + X0[(b * NP + p) * 128 + tid];
    sm[tid] = v; __syncthreads();
    for (int o = 64; o > 0; o >>= 1) { if (tid < o) sm[tid] += sm[tid + o]; __syncthreads(); }
    float mean = sm[0] * (1.f / 128.f); __syncthreads();
    float d = v - mean;
    sm[tid] = d * d; __syncthreads();
    for (int o = 64; o > 0; o >>= 1) { if (tid < o) sm[tid] += sm[tid + o]; __syncthreads(); }
    float rstd = rsqrtf(sm[0] * (1.f / 128.f) + 1e-6f);
    Y[t * 128 + tid] = (v - mean) * rstd * G[tid] + Bt[tid];
}

// ---------------------------------------------------------------------------
// Deformable: attention-weight (128->16, softmax over NS=4) and sampling
// offsets (128->32, tanh). One token per 64-thread block.
// ---------------------------------------------------------------------------
__global__ __launch_bounds__(64) void awso_k(
    const float* __restrict__ XN,
    const float* __restrict__ aw_w, const float* __restrict__ aw_b,
    const float* __restrict__ so_w, const float* __restrict__ so_b,
    float* __restrict__ Wout, float* __restrict__ Off)
{
    __shared__ float sw[16];
    long t = blockIdx.x;
    int j = threadIdx.x;
    const float* x = XN + t * 128;
    if (j < 16) {
        float s = aw_b[j];
        for (int k = 0; k < 128; ++k) s += x[k] * aw_w[k * 16 + j];
        sw[j] = s;
    } else if (j < 48) {
        int jj = j - 16;
        float s = so_b[jj];
        for (int k = 0; k < 128; ++k) s += x[k] * so_w[k * 32 + jj];
        Off[t * 32 + jj] = tanhf(s);
    }
    __syncthreads();
    if (j < 16) {
        int g = j & ~3;
        float mx = fmaxf(fmaxf(sw[g], sw[g + 1]), fmaxf(sw[g + 2], sw[g + 3]));
        float den = __expf(sw[g] - mx) + __expf(sw[g + 1] - mx) +
                    __expf(sw[g + 2] - mx) + __expf(sw[g + 3] - mx);
        Wout[t * 16 + j] = __expf(sw[j] - mx) / den;
    }
}

// ---------------------------------------------------------------------------
// Deformable sampling + per-level embed (C->32) + weighted accumulation.
// One (b,p) per 128-thread block for a given level. Adds into XL (residual).
// ---------------------------------------------------------------------------
__global__ __launch_bounds__(128) void deform_sample_k(
    const float* __restrict__ F, const float* __restrict__ Ref,
    const float* __restrict__ Wgt, const float* __restrict__ Off,
    const float* __restrict__ EW, const float* __restrict__ EB,
    float* XL, int C, int H, int W, int level)
{
    extern __shared__ float fsm[];      // C floats (bilinear feature vector)
    __shared__ float red[128];
    __shared__ float acc[128];
    int bp = blockIdx.x;
    int b = bp / NP, p = bp % NP;
    long tt = ((long)b * 4 + level) * NP + p;
    float rx = Ref[((long)b * NP + p) * 2 + 0];
    float ry = Ref[((long)b * NP + p) * 2 + 1];
    int tid = threadIdx.x;
    acc[tid] = 0.f;
    for (int hs = 0; hs < 16; ++hs) {
        __syncthreads();
        float ox = Off[tt * 32 + hs * 2 + 0] + rx;
        float oy = Off[tt * 32 + hs * 2 + 1] + ry;
        float gx = fminf(fmaxf((ox + 1.f) * 0.5f * (float)(W - 1), 0.f), (float)(W - 1));
        float gy = fminf(fmaxf((oy + 1.f) * 0.5f * (float)(H - 1), 0.f), (float)(H - 1));
        float x0f = floorf(gx), y0f = floorf(gy);
        float wx = gx - x0f, wy = gy - y0f;
        int x0 = (int)x0f, y0 = (int)y0f;
        int x1 = min(x0 + 1, W - 1), y1 = min(y0 + 1, H - 1);
        x0 = min(max(x0, 0), W - 1); y0 = min(max(y0, 0), H - 1);
        for (int cc = tid; cc < C; cc += 128) {
            const float* fc = F + ((long)b * C + cc) * (long)(H * W);
            float v00 = fc[y0 * W + x0], v01 = fc[y0 * W + x1];
            float v10 = fc[y1 * W + x0], v11 = fc[y1 * W + x1];
            fsm[cc] = v00 * (1.f - wy) * (1.f - wx) + v01 * (1.f - wy) * wx +
                      v10 * wy * (1.f - wx) + v11 * wy * wx;
        }
        __syncthreads();
        int e = tid & 31, part = tid >> 5;
        float s = 0.f;
        for (int cc = part; cc < C; cc += 4) s += fsm[cc] * EW[cc * 32 + e];
        red[tid] = s;
        __syncthreads();
        if (part == 0) {
            float tot = red[e] + red[e + 32] + red[e + 64] + red[e + 96] + EB[e];
            int h = hs >> 2;
            acc[h * 32 + e] += Wgt[tt * 16 + hs] * tot;
        }
    }
    __syncthreads();
    XL[tt * 128 + tid] += acc[tid];
}

// ---------------------------------------------------------------------------
// Context-feature init: grid_sample feats[level] at ref, embed C->128,
// add positional embedding, write XL level slice.
// ---------------------------------------------------------------------------
__global__ __launch_bounds__(128) void ctxfeat_k(
    const float* __restrict__ F, const float* __restrict__ Ref,
    const float* __restrict__ FW, const float* __restrict__ FB,
    const float* __restrict__ Pos, float* __restrict__ XL,
    int C, int H, int W, int level)
{
    extern __shared__ float fsm[];
    int bp = blockIdx.x;
    int b = bp / NP, p = bp % NP;
    float ox = Ref[((long)b * NP + p) * 2 + 0];
    float oy = Ref[((long)b * NP + p) * 2 + 1];
    float gx = fminf(fmaxf((ox + 1.f) * 0.5f * (float)(W - 1), 0.f), (float)(W - 1));
    float gy = fminf(fmaxf((oy + 1.f) * 0.5f * (float)(H - 1), 0.f), (float)(H - 1));
    float x0f = floorf(gx), y0f = floorf(gy);
    float wx = gx - x0f, wy = gy - y0f;
    int x0 = (int)x0f, y0 = (int)y0f;
    int x1 = min(x0 + 1, W - 1), y1 = min(y0 + 1, H - 1);
    x0 = min(max(x0, 0), W - 1); y0 = min(max(y0, 0), H - 1);
    for (int cc = threadIdx.x; cc < C; cc += 128) {
        const float* fc = F + ((long)b * C + cc) * (long)(H * W);
        float v00 = fc[y0 * W + x0], v01 = fc[y0 * W + x1];
        float v10 = fc[y1 * W + x0], v11 = fc[y1 * W + x1];
        fsm[cc] = v00 * (1.f - wy) * (1.f - wx) + v01 * (1.f - wy) * wx +
                  v10 * wy * (1.f - wx) + v11 * wy * wx;
    }
    __syncthreads();
    int j = threadIdx.x;
    float s = FB[j];
    for (int cc = 0; cc < C; ++cc) s += fsm[cc] * FW[cc * 128 + j];
    s += Pos[((level + 1) * NP + p) * 128 + j];
    XL[(((long)b * 4 + level) * NP + p) * 128 + j] = s;
}

// X0 init: coord embed of keypoints_2d + pos level 0.
__global__ __launch_bounds__(128) void x0_init_k(
    const float* __restrict__ KP2, const float* __restrict__ CW,
    const float* __restrict__ CB, const float* __restrict__ Pos,
    float* __restrict__ X0)
{
    int bp = blockIdx.x;
    int p = bp % NP;
    int j = threadIdx.x;
    const float* kp = KP2 + (long)bp * 2;
    X0[(long)bp * 128 + j] = kp[0] * CW[j] + kp[1] * CW[128 + j] + CB[j]
                           + Pos[p * 128 + j];
}

// Time embedding MLP (sin/cos -> 128 -> gelu(256) -> 128), one b per block.
__global__ __launch_bounds__(256) void temb_k(
    const float* __restrict__ T,
    const float* __restrict__ W1, const float* __restrict__ B1,
    const float* __restrict__ W2, const float* __restrict__ B2,
    float* __restrict__ TEMB)
{
    __shared__ float e[128];
    __shared__ float h[256];
    int b = blockIdx.x;
    int j = threadIdx.x;
    float t = T[b];
    if (j < 64) {
        float fr = __expf(-(float)j * (logf(10000.f) / 63.f));
        e[j]      = sinf(t * fr);
        e[j + 64] = cosf(t * fr);
    }
    __syncthreads();
    float s = B1[j];
    for (int k = 0; k < 128; ++k) s += e[k] * W1[k * 256 + j];
    h[j] = gelu_f(s);
    __syncthreads();
    if (j < 128) {
        float s2 = B2[j];
        for (int k = 0; k < 256; ++k) s2 += h[k] * W2[k * 128 + j];
        TEMB[(long)b * 128 + j] = s2;
    }
}

// X3D init: coord3d embed + broadcast time embedding.
__global__ __launch_bounds__(128) void x3d_init_k(
    const float* __restrict__ KP3, const float* __restrict__ CW,
    const float* __restrict__ CB, const float* __restrict__ TEMB,
    float* __restrict__ X3D)
{
    int bp = blockIdx.x;
    int b = bp / NP;
    int j = threadIdx.x;
    const float* kp = KP3 + (long)bp * 3;
    X3D[(long)bp * 128 + j] = CB[j] + kp[0] * CW[j] + kp[1] * CW[128 + j]
                            + kp[2] * CW[256 + j] + TEMB[(long)b * 128 + j];
}

// ---------------------------------------------------------------------------
// Softmax attention over n=17 tokens; one (b,head) per 128-thread block.
// Reads f32 QKV, writes f16 AO (feeds proj GEMM).
// ---------------------------------------------------------------------------
__global__ __launch_bounds__(128) void attention_k(
    const float* __restrict__ QKV, _Float16* __restrict__ AO,
    int d, int nh, int hd)
{
    extern __shared__ float sm[];       // Q | K | V : each 17*hd
    __shared__ float Sc[32];
    int bh = blockIdx.x;
    int h = bh % nh, b = bh / nh;
    float* Qs = sm;
    float* Ks = sm + NP * hd;
    float* Vs = sm + 2 * NP * hd;
    for (int i = threadIdx.x; i < NP * hd; i += 128) {
        int n = i / hd, e = i % hd;
        long base = ((long)(b * NP + n)) * 3 * d + h * hd + e;
        Qs[i] = QKV[base];
        Ks[i] = QKV[base + d];
        Vs[i] = QKV[base + 2 * d];
    }
    __syncthreads();
    float scale = rsqrtf((float)hd);
    for (int n = 0; n < NP; ++n) {
        if (threadIdx.x < NP) {
            float s = 0.f;
            for (int e = 0; e < hd; ++e) s += Qs[n * hd + e] * Ks[threadIdx.x * hd + e];
            Sc[threadIdx.x] = s * scale;
        }
        __syncthreads();
        if (threadIdx.x == 0) {
            float mx = Sc[0];
            for (int i = 1; i < NP; ++i) mx = fmaxf(mx, Sc[i]);
            float sum = 0.f;
            for (int i = 0; i < NP; ++i) { float ee = __expf(Sc[i] - mx); Sc[i] = ee; sum += ee; }
            float inv = 1.f / sum;
            for (int i = 0; i < NP; ++i) Sc[i] *= inv;
        }
        __syncthreads();
        for (int e = threadIdx.x; e < hd; e += 128) {
            float o = 0.f;
            for (int m = 0; m < NP; ++m) o += Sc[m] * Vs[m * hd + e];
            AO[((long)(b * NP + n)) * d + h * hd + e] = (_Float16)o;
        }
        __syncthreads();
    }
}

// Assemble FULL = concat(ctx levels 0..4, x3d) per token.
__global__ __launch_bounds__(256) void full_init_k(
    const float* __restrict__ X0, const float* __restrict__ XL,
    const float* __restrict__ X3D, float* __restrict__ FULL)
{
    int bp = blockIdx.x;
    int b = bp / NP, p = bp % NP;
    for (int j = threadIdx.x; j < DJNT; j += 256) {
        int l = j >> 7, e = j & 127;
        float v;
        if (l == 0)      v = X0[(long)bp * 128 + e];
        else if (l <= 4) v = XL[(((long)b * 4 + (l - 1)) * NP + p) * 128 + e];
        else             v = X3D[(long)bp * 128 + e];
        FULL[(long)bp * DJNT + j] = v;
    }
}

// Head: 768 -> 3, one thread per output element.
__global__ void head_k(const float* __restrict__ XN, const float* __restrict__ HW,
                       const float* __restrict__ HB, float* __restrict__ OUT, int total)
{
    int i = blockIdx.x * 256 + threadIdx.x;
    if (i >= total) return;
    int o = i % 3;
    long r = i / 3;
    const float* x = XN + r * DJNT;
    float s = HB[o];
    for (int k = 0; k < DJNT; ++k) s += x[k] * HW[k * 3 + o];
    OUT[i] = s;
}

// ---------------------------------------------------------------------------
// Host side
// ---------------------------------------------------------------------------

// Input index map. d_in[0..7] = arrays in setup_inputs insertion order;
// params pytree flattened JAX-style (dict keys sorted, lists in order).
enum {
    I_KP2 = 0, I_REF, I_KP3, I_F0, I_F1, I_F2, I_F3, I_T,
    P_COORD_B = 8, P_COORD_W, P_C3D_B, P_C3D_W,
    P_CTX0  = 12,   // + blk*20
    P_FEAT0 = 92,   // + lvl*2 (b,w)
    P_HEAD_B = 100, P_HEAD_W, P_HLN_B, P_HLN_G,
    P_JOINT0 = 104, // + blk*12
    P_POS = 152,
    P_RES0 = 153,   // + blk*12
    P_T1_B = 201, P_T1_W, P_T2_B, P_T2_W
};
enum { L_FC1B = 0, L_FC1W, L_FC2B, L_FC2W, L_LN1B, L_LN1G, L_LN2B, L_LN2G,
       L_PRJB, L_PRJW, L_QKVB, L_QKVW };
enum { C_AWB = 0, C_AWW, C_E0B, C_E0W, C_E1B, C_E1W, C_E2B, C_E2W, C_E3B, C_E3W,
       C_FC1B, C_FC1W, C_FC2B, C_FC2W, C_LN1B, C_LN1G, C_LN2B, C_LN2G,
       C_SOB, C_SOW };

static inline void launch_gemm(hipStream_t st, const _Float16* A, const uint32_t* Wp,
                               const float* bias, const float* res,
                               float* Cf, _Float16* Ch,
                               int M, int K, int N, int act) {
    dim3 grid(M / 32, N / 128);
    gemm_wmma_f16<<<grid, 128, 0, st>>>(A, Wp, bias, res, Cf, Ch, M, K, N, act);
}

extern "C" void kernel_launch(void* const* d_in, const int* in_sizes, int n_in,
                              void* d_out, int out_size, void* d_ws, size_t ws_size,
                              hipStream_t stream) {
    (void)in_sizes; (void)n_in; (void)out_size; (void)ws_size;
    #define PF(i) ((const float*)d_in[(i)])

    const int   FC[4]  = {32, 64, 128, 256};
    const int   FH[4]  = {64, 32, 16, 8};
    const int   FWD[4] = {48, 24, 12, 6};
    const float* FEATS[4] = { PF(I_F0), PF(I_F1), PF(I_F2), PF(I_F3) };

    // ---- workspace bump allocator ----
    uint8_t* ws = (uint8_t*)d_ws;
    size_t off = 0;
    auto alloc = [&](size_t bytes) -> void* {
        void* p = ws + off;
        off += (bytes + 255) & ~(size_t)255;
        return p;
    };
    float* X0    = (float*)alloc((size_t)B17 * 128 * 4);
    float* XL    = (float*)alloc((size_t)TL * 128 * 4);
    float* X3D   = (float*)alloc((size_t)B17 * 128 * 4);
    float* TEMB  = (float*)alloc((size_t)NB * 128 * 4);
    float* FULL  = (float*)alloc((size_t)B17 * DJNT * 4);
    float* XN    = (float*)alloc((size_t)B17 * DJNT * 4);   // also f16 alias
    float* AO    = (float*)alloc((size_t)B17 * DJNT * 4);
    float* QKVH  = (float*)alloc((size_t)B17 * 2304 * 4);   // qkv f32 / hidden f16
    float* W16   = (float*)alloc((size_t)TL * 16 * 4);
    float* OFF32 = (float*)alloc((size_t)TL * 32 * 4);

    _Float16* XN16 = (_Float16*)XN;   // f16 activations feeding WMMA GEMMs
    _Float16* AO16 = (_Float16*)AO;
    _Float16* H16  = (_Float16*)QKVH;

    // ---- pack all GEMM weights to f16 fragment layout ----
    auto pack = [&](const float* W, int K, int N) -> uint32_t* {
        uint32_t* dst = (uint32_t*)alloc((size_t)K * N * 2);
        long total = (long)K * N / 2;
        pack_weight_k<<<(int)((total + 255) / 256), 256, 0, stream>>>(W, dst, K, N);
        return dst;
    };
    uint32_t *Wres[4][4], *Wjnt[4][4], *Wctx[4][2];
    for (int d = 0; d < 4; ++d) {
        int rb = P_RES0 + d * 12, jb = P_JOINT0 + d * 12, cb = P_CTX0 + d * 20;
        Wres[d][0] = pack(PF(rb + L_QKVW), 128, 384);
        Wres[d][1] = pack(PF(rb + L_PRJW), 128, 128);
        Wres[d][2] = pack(PF(rb + L_FC1W), 128, 256);
        Wres[d][3] = pack(PF(rb + L_FC2W), 256, 128);
        Wjnt[d][0] = pack(PF(jb + L_QKVW), 768, 2304);
        Wjnt[d][1] = pack(PF(jb + L_PRJW), 768, 768);
        Wjnt[d][2] = pack(PF(jb + L_FC1W), 768, 1536);
        Wjnt[d][3] = pack(PF(jb + L_FC2W), 1536, 768);
        Wctx[d][0] = pack(PF(cb + C_FC1W), 128, 256);
        Wctx[d][1] = pack(PF(cb + C_FC2W), 256, 128);
    }

    // ---- preprocessing ----
    temb_k<<<NB, 256, 0, stream>>>(PF(I_T), PF(P_T1_W), PF(P_T1_B),
                                   PF(P_T2_W), PF(P_T2_B), TEMB);
    x0_init_k<<<B17, 128, 0, stream>>>(PF(I_KP2), PF(P_COORD_W), PF(P_COORD_B),
                                       PF(P_POS), X0);
    x3d_init_k<<<B17, 128, 0, stream>>>(PF(I_KP3), PF(P_C3D_W), PF(P_C3D_B),
                                        TEMB, X3D);
    for (int l = 0; l < 4; ++l)
        ctxfeat_k<<<B17, 128, FC[l] * 4, stream>>>(
            FEATS[l], PF(I_REF), PF(P_FEAT0 + 2 * l + 1), PF(P_FEAT0 + 2 * l),
            PF(P_POS), XL, FC[l], FH[l], FWD[l], l);

    // ---- parallel trunks: deformable ctx blocks + res blocks ----
    for (int d = 0; d < 4; ++d) {
        int cb = P_CTX0 + d * 20;
        // xn = ln1(xl + x0) (f32, feeds offset/weight heads)
        ln_add0_k<<<TL, 128, 0, stream>>>(XL, X0, PF(cb + C_LN1G), PF(cb + C_LN1B), XN);
        awso_k<<<TL, 64, 0, stream>>>(XN, PF(cb + C_AWW), PF(cb + C_AWB),
                                      PF(cb + C_SOW), PF(cb + C_SOB), W16, OFF32);
        for (int l = 0; l < 4; ++l)
            deform_sample_k<<<B17, 128, FC[l] * 4, stream>>>(
                FEATS[l], PF(I_REF), W16, OFF32,
                PF(cb + C_E0W + 2 * l), PF(cb + C_E0B + 2 * l),
                XL, FC[l], FH[l], FWD[l], l);
        // MLP: xl += fc2(gelu(fc1(ln2(xl))))
        layernorm16_k<<<TL, 128, 0, stream>>>(XL, PF(cb + C_LN2G), PF(cb + C_LN2B), XN16, 128);
        launch_gemm(stream, XN16, Wctx[d][0], PF(cb + C_FC1B), nullptr, nullptr, H16, TL, 128, 256, 1);
        launch_gemm(stream, H16, Wctx[d][1], PF(cb + C_FC2B), XL, XL, nullptr, TL, 256, 128, 0);

        // res transformer block on X3D (d=128, 8 heads)
        int rb = P_RES0 + d * 12;
        layernorm16_k<<<B17, 128, 0, stream>>>(X3D, PF(rb + L_LN1G), PF(rb + L_LN1B), XN16, 128);
        launch_gemm(stream, XN16, Wres[d][0], PF(rb + L_QKVB), nullptr, QKVH, nullptr, B17, 128, 384, 0);
        attention_k<<<NB * 8, 128, 3 * NP * 16 * 4, stream>>>(QKVH, AO16, 128, 8, 16);
        launch_gemm(stream, AO16, Wres[d][1], PF(rb + L_PRJB), X3D, X3D, nullptr, B17, 128, 128, 0);
        layernorm16_k<<<B17, 128, 0, stream>>>(X3D, PF(rb + L_LN2G), PF(rb + L_LN2B), XN16, 128);
        launch_gemm(stream, XN16, Wres[d][2], PF(rb + L_FC1B), nullptr, nullptr, H16, B17, 128, 256, 1);
        launch_gemm(stream, H16, Wres[d][3], PF(rb + L_FC2B), X3D, X3D, nullptr, B17, 256, 128, 0);
    }

    // ---- joint transformer (d=768, 8 heads) ----
    full_init_k<<<B17, 256, 0, stream>>>(X0, XL, X3D, FULL);
    for (int d = 0; d < 4; ++d) {
        int jb = P_JOINT0 + d * 12;
        layernorm16_k<<<B17, 128, 0, stream>>>(FULL, PF(jb + L_LN1G), PF(jb + L_LN1B), XN16, DJNT);
        launch_gemm(stream, XN16, Wjnt[d][0], PF(jb + L_QKVB), nullptr, QKVH, nullptr, B17, 768, 2304, 0);
        attention_k<<<NB * 8, 128, 3 * NP * 96 * 4, stream>>>(QKVH, AO16, 768, 8, 96);
        launch_gemm(stream, AO16, Wjnt[d][1], PF(jb + L_PRJB), FULL, FULL, nullptr, B17, 768, 768, 0);
        layernorm16_k<<<B17, 128, 0, stream>>>(FULL, PF(jb + L_LN2G), PF(jb + L_LN2B), XN16, DJNT);
        launch_gemm(stream, XN16, Wjnt[d][2], PF(jb + L_FC1B), nullptr, nullptr, H16, B17, 768, 1536, 1);
        launch_gemm(stream, H16, Wjnt[d][3], PF(jb + L_FC2B), FULL, FULL, nullptr, B17, 1536, 768, 0);
    }

    // ---- head ----
    layernorm_k<<<B17, 128, 0, stream>>>(FULL, PF(P_HLN_G), PF(P_HLN_B), XN, DJNT);
    int total = B17 * 3;
    head_k<<<(total + 255) / 256, 256, 0, stream>>>(XN, PF(P_HEAD_W), PF(P_HEAD_B),
                                                    (float*)d_out, total);
    #undef PF
}
